// M_15178414424540
// MI455X (gfx1250) — matrix operand order repaired and checked by reading, not compiled, not run on hardware
//
#include <hip/hip_runtime.h>

#define S_LEN 8192
#define D_DIM 128
#define BM 64          // query rows per workgroup (4 waves x 16 rows)
#define BN 64          // keys per iteration
#define NWAVES 4
#define VSTR 72        // padded bf16 stride for LDS VT tile (144B = 16B-multiple)

typedef __attribute__((ext_vector_type(16))) __bf16    v16bf;
typedef __attribute__((ext_vector_type(8)))  float     v8f;
typedef __attribute__((ext_vector_type(4)))  float     f32x4;
typedef __attribute__((ext_vector_type(4)))  unsigned  u32x4;
typedef __attribute__((ext_vector_type(2)))  unsigned  u32x2;

union BFrag {
    v16bf v;
    u32x4 q[2];
    unsigned short h[16];
};

__device__ __forceinline__ unsigned short bfbits(float f) {
    unsigned u = __builtin_bit_cast(unsigned, f);
    u += 0x7FFFu + ((u >> 16) & 1u);          // round-to-nearest-even
    return (unsigned short)(u >> 16);
}

// ---------------- prep 1: Qb = bf16(Q*scale), Kb = bf16(K), row-major ----------------
__global__ __launch_bounds__(256) void prep_qk(
    const float* __restrict__ Q, const float* __restrict__ K,
    unsigned short* __restrict__ Qb, unsigned short* __restrict__ Kb)
{
    const float scale = 0.08838834764831845f;   // 1/sqrt(128)
    const int i = blockIdx.x * 256 + threadIdx.x;        // over S*D/4 float4s
    f32x4 q = ((const f32x4*)Q)[i];
    f32x4 k = ((const f32x4*)K)[i];
    u32x2 pq, pk;
    pq.x = (unsigned)bfbits(q[0] * scale) | ((unsigned)bfbits(q[1] * scale) << 16);
    pq.y = (unsigned)bfbits(q[2] * scale) | ((unsigned)bfbits(q[3] * scale) << 16);
    pk.x = (unsigned)bfbits(k[0]) | ((unsigned)bfbits(k[1]) << 16);
    pk.y = (unsigned)bfbits(k[2]) | ((unsigned)bfbits(k[3]) << 16);
    ((u32x2*)Qb)[i] = pq;
    ((u32x2*)Kb)[i] = pk;
}

// ---------------- prep 2: VTb[d][s] = bf16(V[s][d]) (tiled LDS transpose) ----------------
__global__ __launch_bounds__(256) void prep_vt(
    const float* __restrict__ V, unsigned short* __restrict__ VTb)
{
    __shared__ __align__(16) unsigned short t[D_DIM * 64];   // [d][s] 16 KB
    const int tid = threadIdx.x;
    const int s0  = blockIdx.x * 64;
#pragma unroll
    for (int i = 0; i < 8; ++i) {
        const int idx4 = i * 256 + tid;          // 64 rows * 32 float4
        const int s  = idx4 >> 5;
        const int dc = (idx4 & 31) * 4;
        f32x4 v = *(const f32x4*)(V + (size_t)(s0 + s) * D_DIM + dc);
#pragma unroll
        for (int c = 0; c < 4; ++c) t[(dc + c) * 64 + s] = bfbits(v[c]);
    }
    __syncthreads();
#pragma unroll
    for (int i = 0; i < 4; ++i) {
        const int idx = i * 256 + tid;           // 1024 uint4s: d = idx/8, chunk = idx%8
        const int d = idx >> 3, c = idx & 7;
        *(u32x4*)(VTb + (size_t)d * S_LEN + s0 + c * 8) = *(const u32x4*)&t[d * 64 + c * 8];
    }
}

// ---- tile pipeline helpers: global -> regs (prefetch), regs -> LDS (commit) ----
__device__ __forceinline__ void load_tile_regs(
    const unsigned short* __restrict__ Kb, const unsigned short* __restrict__ VTb,
    int kb, int tid, u32x4 kreg[8], u32x4 vreg[8])
{
    const u32x4* src = (const u32x4*)(Kb + (size_t)kb * D_DIM);   // contiguous 16 KB tile
#pragma unroll
    for (int i = 0; i < 8; ++i) kreg[i] = src[i * 128 + tid];
#pragma unroll
    for (int i = 0; i < 8; ++i) {
        const int idx = i * 128 + tid;           // 1024 uint4s: d = idx/8, chunk = idx%8
        const int d = idx >> 3, c = idx & 7;
        vreg[i] = *(const u32x4*)(VTb + (size_t)d * S_LEN + kb + c * 8);
    }
}

__device__ __forceinline__ void store_tile_lds(
    unsigned short* sK, unsigned short* sVT, int tid,
    const u32x4 kreg[8], const u32x4 vreg[8])
{
#pragma unroll
    for (int i = 0; i < 8; ++i) ((u32x4*)sK)[i * 128 + tid] = kreg[i];
#pragma unroll
    for (int i = 0; i < 8; ++i) {
        const int idx = i * 128 + tid;
        const int d = idx >> 3, c = idx & 7;
        *(u32x4*)&sVT[d * VSTR + c * 8] = vreg[i];
    }
}

// ---------------- main flash-attention kernel (bf16 WMMA, fp32 accum) ----------------
__global__ __launch_bounds__(128) void fa_fwd_bf16_wmma(
    const unsigned short* __restrict__ Qb, const unsigned short* __restrict__ Kb,
    const unsigned short* __restrict__ VTb, float* __restrict__ O)
{
    __shared__ __align__(16) unsigned short sK[BN * D_DIM];       // 16 KB  [key][d]
    __shared__ __align__(16) unsigned short sVT[D_DIM * VSTR];    // 18 KB  [d][key]
    __shared__ __align__(16) unsigned short sP[NWAVES * 16 * BN]; //  8 KB per-wave P

    const int tid  = threadIdx.x;
    const int wave = tid >> 5;
    const int lane = tid & 31;
    const int l16  = lane & 15;
    const int half = lane >> 4;

    const int   qrow0 = blockIdx.x * BM + wave * 16;
    const float LOG2E = 1.4426950408889634f;

    // ---- Q A-fragments: straight bf16 b128 loads ----
    BFrag qa[4];
    {
        const unsigned short* qp = Qb + (size_t)(qrow0 + l16) * D_DIM;
#pragma unroll
        for (int t = 0; t < 4; ++t) {
            const int d0 = 32 * t + 8 * half;
            qa[t].q[0] = *(const u32x4*)(qp + d0);        // K = d0..d0+7
            qa[t].q[1] = *(const u32x4*)(qp + d0 + 16);   // K = d0+16..d0+23
        }
    }

    v8f acc[8];
#pragma unroll
    for (int i = 0; i < 8; ++i) acc[i] = (v8f){0.f,0.f,0.f,0.f,0.f,0.f,0.f,0.f};
    float mrow[8], lrow[8];
#pragma unroll
    for (int j = 0; j < 8; ++j) { mrow[j] = -3.0e38f; lrow[j] = 0.f; }

    unsigned short* pbase = &sP[wave * 16 * BN];

    // software pipeline: tile 0 into regs before the loop
    u32x4 kreg[8], vreg[8];
    load_tile_regs(Kb, VTb, 0, tid, kreg, vreg);

    for (int kb = 0; kb < S_LEN; kb += BN) {
        __syncthreads();                       // previous tile fully consumed
        store_tile_lds(sK, sVT, tid, kreg, vreg);
        __syncthreads();                       // tile visible to all waves

        // issue next tile's global loads now; they complete under the WMMAs below
        if (kb + BN < S_LEN)
            load_tile_regs(Kb, VTb, kb + BN, tid, kreg, vreg);
        // L2 prefetch two tiles ahead (global_prefetch_b8)
        if (kb + 2 * BN < S_LEN) {
            __builtin_prefetch(Kb + (size_t)(kb + 2 * BN) * D_DIM + tid * 64, 0, 1);
            __builtin_prefetch(VTb + (size_t)tid * S_LEN + kb + 2 * BN, 0, 1);
        }

        // ---- S = Q K^T : 4 n-tiles x (K=128/32) WMMA, fragments preloaded ----
        v8f sc[4];
#pragma unroll
        for (int nt = 0; nt < 4; ++nt) {
            BFrag kf[4];
#pragma unroll
            for (int t = 0; t < 4; ++t) {      // 8 ds_load_b128 issued up front
                const unsigned short* kp = &sK[(nt * 16 + l16) * D_DIM + 32 * t + 8 * half];
                kf[t].q[0] = *(const u32x4*)kp;
                kf[t].q[1] = *(const u32x4*)(kp + 16);
            }
            v8f c = (v8f){0.f,0.f,0.f,0.f,0.f,0.f,0.f,0.f};
#pragma unroll
            for (int t = 0; t < 4; ++t)
                c = __builtin_amdgcn_wmma_f32_16x16x32_bf16(
                        false, qa[t].v, false, kf[t].v, (short)0, c, false, false);
            sc[nt] = c;
        }

        // ---- online softmax (row = j + 8*half, N = l16) ----
        float alpha[8];
#pragma unroll
        for (int j = 0; j < 8; ++j) {
            float t = fmaxf(fmaxf(sc[0][j], sc[1][j]), fmaxf(sc[2][j], sc[3][j]));
            t = fmaxf(t, __shfl_xor(t, 1, 32));
            t = fmaxf(t, __shfl_xor(t, 2, 32));
            t = fmaxf(t, __shfl_xor(t, 4, 32));
            t = fmaxf(t, __shfl_xor(t, 8, 32));
            const float mnew = fmaxf(mrow[j], t);
            alpha[j] = __builtin_exp2f((mrow[j] - mnew) * LOG2E);
            mrow[j]  = mnew;
        }
        float rsum[8];
#pragma unroll
        for (int j = 0; j < 8; ++j) rsum[j] = 0.f;
#pragma unroll
        for (int nt = 0; nt < 4; ++nt) {
#pragma unroll
            for (int j = 0; j < 8; ++j) {
                const float p = __builtin_exp2f((sc[nt][j] - mrow[j]) * LOG2E);
                rsum[j] += p;
                pbase[(j + 8 * half) * BN + nt * 16 + l16] = bfbits(p);
            }
        }
#pragma unroll
        for (int j = 0; j < 8; ++j) {
            float s = rsum[j];
            s += __shfl_xor(s, 1, 32);
            s += __shfl_xor(s, 2, 32);
            s += __shfl_xor(s, 4, 32);
            s += __shfl_xor(s, 8, 32);
            lrow[j] = lrow[j] * alpha[j] + s;
        }
#pragma unroll
        for (int no = 0; no < 8; ++no)
#pragma unroll
            for (int j = 0; j < 8; ++j) acc[no][j] *= alpha[j];

        // P stores -> P loads are same-wave LDS (in-order); explicit wait for safety
        asm volatile("s_wait_dscnt 0x0" ::: "memory");

        // ---- reload P as A fragments (16x64 -> 2 frags) ----
        BFrag pa[2];
#pragma unroll
        for (int f = 0; f < 2; ++f) {
            const unsigned short* pp = pbase + l16 * BN + f * 32 + 8 * half;
            pa[f].q[0] = *(const u32x4*)pp;          // K = f*32 + 8h + 0..7
            pa[f].q[1] = *(const u32x4*)(pp + 16);   // K = f*32 + 16 + 8h + 0..7
        }

        // ---- O += P V : 8 n-tiles x (K=64/32) WMMA, fragments preloaded ----
#pragma unroll
        for (int no = 0; no < 8; ++no) {
            BFrag vf[2];
#pragma unroll
            for (int f = 0; f < 2; ++f) {      // 4 ds_load_b128 issued up front
                const unsigned short* vp = &sVT[(no * 16 + l16) * VSTR + f * 32 + 8 * half];
                vf[f].q[0] = *(const u32x4*)vp;
                vf[f].q[1] = *(const u32x4*)(vp + 16);
            }
#pragma unroll
            for (int f = 0; f < 2; ++f)
                acc[no] = __builtin_amdgcn_wmma_f32_16x16x32_bf16(
                              false, pa[f].v, false, vf[f].v, (short)0, acc[no], false, false);
        }
    }

    // ---- normalize + store fp32 output ----
#pragma unroll
    for (int j = 0; j < 8; ++j) {
        const float inv = 1.0f / lrow[j];
        const size_t row = (size_t)(qrow0 + j + 8 * half);
#pragma unroll
        for (int no = 0; no < 8; ++no)
            O[row * D_DIM + no * 16 + l16] = acc[no][j] * inv;
    }
}

extern "C" void kernel_launch(void* const* d_in, const int* in_sizes, int n_in,
                              void* d_out, int out_size, void* d_ws, size_t ws_size,
                              hipStream_t stream) {
    const float* q = (const float*)d_in[0];
    const float* k = (const float*)d_in[1];
    const float* v = (const float*)d_in[2];
    float* o = (float*)d_out;

    // workspace layout (bf16): Qb [S*D] | Kb [S*D] | VTb [D*S]  => 6 MB total
    unsigned short* wsb = (unsigned short*)d_ws;
    unsigned short* Qb  = wsb;
    unsigned short* Kb  = wsb + (size_t)S_LEN * D_DIM;
    unsigned short* VTb = wsb + (size_t)2 * S_LEN * D_DIM;

    prep_qk<<<dim3((S_LEN * D_DIM / 4) / 256), dim3(256), 0, stream>>>(q, k, Qb, Kb);
    prep_vt<<<dim3(S_LEN / 64), dim3(256), 0, stream>>>(v, VTb);
    fa_fwd_bf16_wmma<<<dim3(S_LEN / BM), dim3(128), 0, stream>>>(Qb, Kb, VTb, o);
}
